// Decoder_85650237817493
// MI455X (gfx1250) — compile-verified, matrix operand order checked by reading, not run on hardware
//
#include <hip/hip_runtime.h>

#define Vv  32000
#define Mm  512
#define E2c 1024
#define Dc  1024
#define Bc  256
#define Sc  128

typedef __attribute__((ext_vector_type(8)))  float  v8f;
typedef __attribute__((ext_vector_type(16))) __bf16 v16bf;
typedef __attribute__((ext_vector_type(2)))  __bf16 bf16x2;
typedef __attribute__((ext_vector_type(4)))  unsigned u32x4;
typedef __attribute__((ext_vector_type(4)))  int      i32x4;
typedef __attribute__((ext_vector_type(8)))  int      i32x8;

union BF2U { bf16x2 v; unsigned u; };

__device__ __forceinline__ unsigned pack_bf16x2(float x, float y) {
  BF2U c;
  c.v[0] = (__bf16)x;
  c.v[1] = (__bf16)y;
  return c.u;
}

// ---------------------------------------------------------------------------
// Stage a contiguous panel of `elems` fp32 from global into LDS.
// Preferred path: Tensor Data Mover (1-D descriptor), issued by wave 0 and
// fenced with s_wait_tensorcnt. Fallback: cooperative vector copy.
// ---------------------------------------------------------------------------
__device__ __forceinline__ void panel_to_lds(const float* __restrict__ g,
                                             float* stage, int elems) {
#if __has_builtin(__builtin_amdgcn_tensor_load_to_lds)
  if (threadIdx.x < 32) {
    const unsigned long long ga = (unsigned long long)(size_t)(const void*)g;
    const unsigned lds_off = (unsigned)(size_t)(void*)stage;
    u32x4 g0;
    g0[0] = 1u;                                   // count=1, user mode, no gather
    g0[1] = lds_off;                              // lds_addr (bytes)
    g0[2] = (unsigned)(ga & 0xffffffffu);         // global_addr[31:0]
    g0[3] = (unsigned)((ga >> 32) & 0x01ffffffu)  // global_addr[56:32]
            | (2u << 30);                         // type = 2 (image/tensor)
    i32x8 g1;
    g1[0] = (int)(2u << 16);                      // data_size = 4B
    g1[1] = (int)(((unsigned)elems & 0xffffu) << 16);         // tensor_dim0 lo
    g1[2] = (int)((((unsigned)elems >> 16) & 0xffffu)         // tensor_dim0 hi
                  | (1u << 16));                              // tensor_dim1 = 1
    g1[3] = (int)(((unsigned)elems & 0xffffu) << 16);         // tile_dim0 = elems
    g1[4] = 1;                                                // tile_dim1 = 1
    g1[5] = (int)(unsigned)elems;                             // stride0 lo
    g1[6] = 0;                                                // stride0 hi | stride1 lo
    g1[7] = 0;
    i32x4 z4 = {0, 0, 0, 0};
#if __clang_major__ >= 23
    i32x8 z8 = {0, 0, 0, 0, 0, 0, 0, 0};
    __builtin_amdgcn_tensor_load_to_lds(g0, g1, z4, z4, z8, 0);
#else
    __builtin_amdgcn_tensor_load_to_lds(g0, g1, z4, z4, 0);
#endif
    __builtin_amdgcn_s_wait_tensorcnt(0);
  }
#else
  float2* d = (float2*)stage;
  const float2* s = (const float2*)g;
  const int n2 = elems >> 1;
  for (int i = threadIdx.x; i < n2; i += 128) d[i] = s[i];
#endif
}

// A-fragment (16x32 bf16) per CDNA5 ISA register layout:
// lane L: row = L&15, K-base = (L>>4)*8; VGPR j<4 -> K = base+2j; j>=4 -> 16+base+2(j-4)
__device__ __forceinline__ v16bf load_tile_bf16(const float* __restrict__ p, int ld, int lane) {
  const int r  = lane & 15;
  const int kb = (lane >> 4) << 3;
  const float* row = p + r * ld;
  v16bf t;
#pragma unroll
  for (int j = 0; j < 8; ++j) {
    const int koff = ((j & 4) << 2) + kb + ((j & 3) << 1);
    const float2 f = *(const float2*)(row + koff);
    t[2 * j]     = (__bf16)f.x;
    t[2 * j + 1] = (__bf16)f.y;
  }
  return t;
}

// Same layout from LDS holding packed bf16 pairs (row pitch in uints).
__device__ __forceinline__ v16bf load_tile_lds(const unsigned* s, int pitch, int lane) {
  const int r  = lane & 15;
  const int kb = (lane >> 4) << 3;
  v16bf t;
#pragma unroll
  for (int j = 0; j < 8; ++j) {
    const int koff = ((j & 4) << 2) + kb + ((j & 3) << 1);
    BF2U c;
    c.u = s[r * pitch + (koff >> 1)];
    t[2 * j]     = c.v[0];
    t[2 * j + 1] = c.v[1];
  }
  return t;
}

__device__ __forceinline__ v8f wmma_bf16(v16bf a, v16bf b, v8f c) {
  return __builtin_amdgcn_wmma_f32_16x16x32_bf16(false, a, false, b, (short)0, c,
                                                 false, false);
}

// ---------------------------------------------------------------------------
// C[R,N] = X[R,K] @ W[N,K]^T + bias.  4 waves/block, block covers 16 n-tiles
// of one m-tile; the 16xK X panel is TDM-staged to LDS once (fp32), packed to
// bf16 once, then the inner loop is barrier-free: 1 A-fragment feeds 4
// independent WMMAs (hides WMMA->WMMA hazard slack).
// Requires R%16==0, (N/16)%16==0, K%32==0.  Dynamic LDS = 96*K bytes.
// ---------------------------------------------------------------------------
__global__ void __launch_bounds__(128)
wmma_gemm_xwt(const float* __restrict__ X, const float* __restrict__ W,
              const float* __restrict__ bias, float* __restrict__ C,
              int R, int N, int K) {
  extern __shared__ char smem[];
  float*    stage = (float*)smem;                            // 16*K fp32
  unsigned* es    = (unsigned*)(smem + (size_t)16 * K * 4);  // 16*(K/2) packed
  const int lane = threadIdx.x & 31;
  const int wave = threadIdx.x >> 5;
  const int ngrp = (N >> 4) >> 4;                 // groups of 16 n-tiles
  const int mt   = blockIdx.x / ngrp;
  const int nt0  = ((blockIdx.x % ngrp) << 4) + wave;  // wave: nt0,+4,+8,+12

  panel_to_lds(X + mt * 16 * K, stage, 16 * K);   // panel rows are contiguous
  __syncthreads();
  const int pairs = K << 3;                       // 16*K/2
  for (int i = threadIdx.x; i < pairs; i += 128)
    es[i] = pack_bf16x2(stage[2 * i], stage[2 * i + 1]);
  __syncthreads();

  const float* wp0 = W + nt0 * 16 * K;
  const float* wp1 = wp0 + 64 * K;
  const float* wp2 = wp0 + 128 * K;
  const float* wp3 = wp0 + 192 * K;
  const int pitch = K >> 1;
  v8f acc0 = {}, acc1 = {}, acc2 = {}, acc3 = {};
  for (int k0 = 0; k0 < K; k0 += 32) {
    __builtin_prefetch(wp0 + k0 + 64, 0, 1);
    __builtin_prefetch(wp1 + k0 + 64, 0, 1);
    __builtin_prefetch(wp2 + k0 + 64, 0, 1);
    __builtin_prefetch(wp3 + k0 + 64, 0, 1);
    v16bf a  = load_tile_lds(es + (k0 >> 1), pitch, lane);
    v16bf b0 = load_tile_bf16(wp0 + k0, K, lane);
    v16bf b1 = load_tile_bf16(wp1 + k0, K, lane);
    v16bf b2 = load_tile_bf16(wp2 + k0, K, lane);
    v16bf b3 = load_tile_bf16(wp3 + k0, K, lane);
    acc0 = wmma_bf16(a, b0, acc0);
    acc1 = wmma_bf16(a, b1, acc1);
    acc2 = wmma_bf16(a, b2, acc2);
    acc3 = wmma_bf16(a, b3, acc3);
  }

  const int n     = lane & 15;
  const int mBase = (lane >> 4) << 3;
  const float bv0 = bias[nt0 * 16 + n];
  const float bv1 = bias[nt0 * 16 + 64 + n];
  const float bv2 = bias[nt0 * 16 + 128 + n];
  const float bv3 = bias[nt0 * 16 + 192 + n];
  int base = (mt * 16 + mBase) * N + nt0 * 16 + n;   // all outputs < 2^31 elems
#pragma unroll
  for (int j = 0; j < 8; ++j) {
    C[base]       = acc0[j] + bv0;
    C[base + 64]  = acc1[j] + bv1;
    C[base + 128] = acc2[j] + bv2;
    C[base + 192] = acc3[j] + bv3;
    base += N;
  }
}

// ---------------------------------------------------------------------------
// Fused attention scores: TDM-stage the contiguous 16x1024 enc tile, pack to
// bf16, 4 waves split the 64 n-tiles of Ua (2 tiles per wave-iteration, WMMA
// over K), epilogue tanh(acc + proj_h + bUa) * v, shuffle + ds_add_f32
// reduction over N.  Dynamic LDS = 64KB stage + 32KB packed + 64B accum.
// ---------------------------------------------------------------------------
__global__ void __launch_bounds__(128)
attn_scores_kernel(const float* __restrict__ enc, const float* __restrict__ Ua,
                   const float* __restrict__ bUa, const float* __restrict__ ph,
                   const float* __restrict__ vvec, float* __restrict__ scores) {
  extern __shared__ char smem[];
  float*    stage = (float*)smem;                       // 16*1024 fp32
  unsigned* es    = (unsigned*)(smem + 65536);          // 16*512 packed
  float*    sacc  = (float*)(smem + 65536 + 32768);     // 16 row accumulators
  const int lane = threadIdx.x & 31;
  const int wave = threadIdx.x >> 5;
  const int r0   = blockIdx.x << 4;       // flattened (b*S + s); never crosses b
  const int b    = r0 >> 7;

  panel_to_lds(enc + r0 * E2c, stage, 16 * E2c);  // 16 consecutive rows: contiguous
  if (threadIdx.x < 16) sacc[threadIdx.x] = 0.f;
  __syncthreads();
  for (int i = threadIdx.x; i < 16 * 512; i += 128)
    es[i] = pack_bf16x2(stage[2 * i], stage[2 * i + 1]);
  __syncthreads();

  const int n     = lane & 15;
  const int mBase = (lane >> 4) << 3;
  float part[8];
#pragma unroll
  for (int j = 0; j < 8; ++j) part[j] = 0.f;

  for (int nt = wave; nt < (Dc >> 4); nt += 8) {  // this iter: nt and nt+4
    const float* wp0 = Ua + nt * 16 * E2c;
    const float* wp1 = wp0 + 64 * E2c;
    v8f acc0 = {}, acc1 = {};
    for (int k0 = 0; k0 < E2c; k0 += 32) {
      __builtin_prefetch(wp0 + k0 + 64, 0, 1);
      __builtin_prefetch(wp1 + k0 + 64, 0, 1);
      v16bf a  = load_tile_lds(es + (k0 >> 1), 512, lane);
      v16bf b0 = load_tile_bf16(wp0 + k0, E2c, lane);
      v16bf b1 = load_tile_bf16(wp1 + k0, E2c, lane);
      acc0 = wmma_bf16(a, b0, acc0);
      acc1 = wmma_bf16(a, b1, acc1);
    }
    const int ng0 = (nt << 4) + n;
    const int ng1 = ng0 + 64;
    const float add0 = bUa[ng0] + ph[b * Dc + ng0];
    const float add1 = bUa[ng1] + ph[b * Dc + ng1];
    const float vv0  = vvec[ng0];
    const float vv1  = vvec[ng1];
#pragma unroll
    for (int j = 0; j < 8; ++j) {
      part[j] += tanhf(acc0[j] + add0) * vv0;
      part[j] += tanhf(acc1[j] + add1) * vv1;
    }
  }

#pragma unroll
  for (int j = 0; j < 8; ++j) {      // reduce over the 16 N-lanes of each half
    float s = part[j];
    s += __shfl_xor(s, 1, 32);
    s += __shfl_xor(s, 2, 32);
    s += __shfl_xor(s, 4, 32);
    s += __shfl_xor(s, 8, 32);
    if (n == 0) atomicAdd(&sacc[mBase + j], s);   // ds_add_f32
  }
  __syncthreads();
  if (threadIdx.x < 16) scores[r0 + threadIdx.x] = sacc[threadIdx.x];
}

__global__ void __launch_bounds__(128)
softmax_kernel(const float* __restrict__ sc, float* __restrict__ attn) {
  __shared__ float redm[4];
  __shared__ float reds[4];
  const int b    = blockIdx.x;
  const int lane = threadIdx.x & 31;
  const int wave = threadIdx.x >> 5;
  const float x = sc[b * Sc + threadIdx.x];
  float m = x;
#pragma unroll
  for (int off = 16; off; off >>= 1) m = fmaxf(m, __shfl_xor(m, off, 32));
  if (lane == 0) redm[wave] = m;
  __syncthreads();
  m = fmaxf(fmaxf(redm[0], redm[1]), fmaxf(redm[2], redm[3]));
  const float e = __expf(x - m);
  float s = e;
#pragma unroll
  for (int off = 16; off; off >>= 1) s += __shfl_xor(s, off, 32);
  if (lane == 0) reds[wave] = s;
  __syncthreads();
  s = reds[0] + reds[1] + reds[2] + reds[3];
  attn[b * Sc + threadIdx.x] = e / s;
}

__global__ void __launch_bounds__(256)
context_embed_kernel(const float* __restrict__ attn, const float* __restrict__ enc,
                     const float* __restrict__ emb, const int* __restrict__ x,
                     float* __restrict__ rnn_in) {
  __shared__ float aw[Sc];
  const int b = blockIdx.x;
  const int t = threadIdx.x;
  if (t < Sc) aw[t] = attn[b * Sc + t];
  __syncthreads();
  const int tok = x[b];
  for (int c = t; c < Mm; c += 256)
    rnn_in[b * (Mm + E2c) + c] = emb[tok * Mm + c];
  for (int e = t; e < E2c; e += 256) {
    float acc = 0.f;
    const float* ep = enc + b * Sc * E2c + e;
#pragma unroll 4
    for (int s = 0; s < Sc; ++s) acc += aw[s] * ep[s * E2c];
    rnn_in[b * (Mm + E2c) + Mm + e] = acc;
  }
}

__global__ void __launch_bounds__(256)
gru_combine_kernel(const float* __restrict__ gi, const float* __restrict__ gh,
                   const float* __restrict__ hidden, float* __restrict__ hnew) {
  const int i = blockIdx.x * 256 + threadIdx.x;   // b*D + d
  const int b = i >> 10;
  const int d = i & (Dc - 1);
  const int gb = b * 3 * Dc;
  const float r  = 1.f / (1.f + __expf(-(gi[gb + d] + gh[gb + d])));
  const float z  = 1.f / (1.f + __expf(-(gi[gb + Dc + d] + gh[gb + Dc + d])));
  const float nn = tanhf(gi[gb + 2 * Dc + d] + r * gh[gb + 2 * Dc + d]);
  hnew[i] = (1.f - z) * nn + z * hidden[i];
}

extern "C" void kernel_launch(void* const* d_in, const int* in_sizes, int n_in,
                              void* d_out, int out_size, void* d_ws, size_t ws_size,
                              hipStream_t stream) {
  (void)in_sizes; (void)n_in; (void)out_size; (void)ws_size;
  const int*   x      = (const int*)  d_in[0];
  const float* hidden = (const float*)d_in[1];
  const float* enc    = (const float*)d_in[2];
  const float* emb    = (const float*)d_in[3];
  const float* Wa     = (const float*)d_in[4];
  const float* ba     = (const float*)d_in[5];
  const float* Ua     = (const float*)d_in[6];
  const float* bUa    = (const float*)d_in[7];
  const float* vvec   = (const float*)d_in[8];
  const float* W_ih   = (const float*)d_in[9];
  const float* W_hh   = (const float*)d_in[10];
  const float* b_ih   = (const float*)d_in[11];
  const float* b_hh   = (const float*)d_in[12];
  const float* fc_W   = (const float*)d_in[13];
  const float* fc_b   = (const float*)d_in[14];

  // Outputs, concatenated flat: prediction [B,V], h_new [B,D], attn_w [B,S]
  float* pred = (float*)d_out;
  float* hnew = pred + (size_t)Bc * Vv;
  float* attn = hnew + (size_t)Bc * Dc;

  // Workspace (~9 MB fp32)
  float* ws     = (float*)d_ws;
  float* proj_h = ws;                                // 256*1024
  float* scores = proj_h + Bc * Dc;                  // 256*128
  float* rnn_in = scores + Bc * Sc;                  // 256*1536
  float* gi     = rnn_in + Bc * (Mm + E2c);          // 256*3072
  float* gh     = gi + Bc * 3 * Dc;                  // 256*3072

  const size_t sh1024 = (size_t)96 * 1024;           // 16*K*4 + 16*K/2*4, K=1024
  const size_t sh1536 = (size_t)144 * 1024;          // K=1536
  const size_t shAttn = 65536 + 32768 + 64;

  // 1) proj_h = hidden @ Wa^T + ba        [256,1024], K=1024
  wmma_gemm_xwt<<<(16 * 64) / 16, 128, sh1024, stream>>>(hidden, Wa, ba, proj_h,
                                                         Bc, Dc, Dc);
  // 2) attention scores (enc @ Ua^T fused with tanh/v-dot)   68.7 GFLOP
  attn_scores_kernel<<<(Bc * Sc) / 16, 128, shAttn, stream>>>(enc, Ua, bUa,
                                                              proj_h, vvec, scores);
  // 3) softmax over S -> attn output
  softmax_kernel<<<Bc, Sc, 0, stream>>>(scores, attn);
  // 4) context = attn @ enc, concat with embedding gather
  context_embed_kernel<<<Bc, 256, 0, stream>>>(attn, enc, emb, x, rnn_in);
  // 5) gi = rnn_in @ W_ih^T + b_ih        [256,3072], K=1536
  wmma_gemm_xwt<<<(16 * 192) / 16, 128, sh1536, stream>>>(rnn_in, W_ih, b_ih, gi,
                                                          Bc, 3 * Dc, Mm + E2c);
  // 6) gh = hidden @ W_hh^T + b_hh        [256,3072], K=1024
  wmma_gemm_xwt<<<(16 * 192) / 16, 128, sh1024, stream>>>(hidden, W_hh, b_hh, gh,
                                                          Bc, 3 * Dc, Dc);
  // 7) GRU elementwise combine -> h_new output
  gru_combine_kernel<<<(Bc * Dc) / 256, 256, 0, stream>>>(gi, gh, hidden, hnew);
  // 8) prediction = h_new @ fc_W^T + fc_b [256,32000], K=1024   16.8 GFLOP
  wmma_gemm_xwt<<<(16 * 2000) / 16, 128, sh1024, stream>>>(hnew, fc_W, fc_b, pred,
                                                           Bc, Vv, Dc);
}
